// OuterProductMean_8435315769811
// MI455X (gfx1250) — matrix operand order repaired and checked by reading, not compiled, not run on hardware
//
#include <hip/hip_runtime.h>

typedef __attribute__((ext_vector_type(16))) __bf16 v16bf;
typedef __attribute__((ext_vector_type(8)))  __bf16 v8bf;
typedef __attribute__((ext_vector_type(8)))  float  v8f;

typedef unsigned int u32x4 __attribute__((ext_vector_type(4)));
typedef int          i32x4 __attribute__((ext_vector_type(4)));
typedef int          i32x8 __attribute__((ext_vector_type(8)));

union Frag { v16bf v; v8bf h[2]; };

#if defined(__HIP_DEVICE_COMPILE__)
  #if __has_builtin(__builtin_amdgcn_tensor_load_to_lds) && __has_builtin(__builtin_amdgcn_s_wait_tensorcnt)
    #define OPM_TDM 1
  #elif __has_builtin(__builtin_amdgcn_global_load_async_to_lds_b128) && __has_builtin(__builtin_amdgcn_s_wait_asynccnt)
    #define OPM_ASYNC 1
  #endif
#endif

#define AS1 __attribute__((address_space(1)))
#define AS3 __attribute__((address_space(3)))

#if defined(OPM_TDM)
// ---------------------------------------------------------------------------
// Issue one TDM 2D tile load: 128 rows x 128 bf16 (contiguous rows, row
// stride 128 elems) from global into LDS with +16B padding every 256B row
// (pad_interval=5 -> 32*8B=256B, pad_amount=3 -> 4 DWORDs = 16B), i.e. an
// LDS row stride of 136 bf16 — matching the WMMA fragment addressing below.
// ---------------------------------------------------------------------------
__device__ inline void tdm_load_tile_2d(const __bf16* gsrc, unsigned lds_byte_addr)
{
    unsigned long long ga = (unsigned long long)(const void*)gsrc;
    u32x4 g0;
    g0.x = 1u;                                    // count=1, user mode
    g0.y = lds_byte_addr;                         // LDS dest (bytes)
    g0.z = (unsigned)ga;                          // global_addr[31:0]
    g0.w = (unsigned)((ga >> 32) & 0x01FFFFFFull) // global_addr[56:32]
         | (2u << 30);                            // type = 2 ("image")
    i32x8 g1;
    g1[0] = (int)((1u << 16)                      // data_size = 2 bytes
                | (1u << 20)                      // pad_enable
                | (5u << 22)                      // pad_interval: 256B
                | (3u << 25));                    // pad_amount: 4 DWORDs
    g1[1] = (int)(128u << 16);                    // tensor_dim0 = 128 (lo16)
    g1[2] = (int)(8192u << 16);                   // tensor_dim1 = 8192 (lo16)
    g1[3] = (int)(128u << 16);                    // tile_dim0 = 128
    g1[4] = 128;                                  // tile_dim1 = 128, tile_dim2 = 0
    g1[5] = 128;                                  // tensor_dim0_stride = 128
    g1[6] = 0;
    g1[7] = 0;
    i32x4 z4 = {0, 0, 0, 0};
#if __clang_major__ >= 23
    i32x8 z8 = {0, 0, 0, 0, 0, 0, 0, 0};
    __builtin_amdgcn_tensor_load_to_lds(g0, g1, z4, z4, z8, 0);
#else
    __builtin_amdgcn_tensor_load_to_lds(g0, g1, z4, z4, 0);
#endif
}
#endif

// ---------------------------------------------------------------------------
// Kernel 0a: Wo [1024 x 128] f32  ->  WoT [128 x 1024] bf16 (z-major)
// ---------------------------------------------------------------------------
__global__ __launch_bounds__(256) void prep_wot_kernel(
    const float* __restrict__ Wo, __bf16* __restrict__ WoT)
{
    int idx = blockIdx.x * 256 + threadIdx.x;     // 131072 = 128*1024
    int z = idx >> 10;
    int k = idx & 1023;
    WoT[idx] = (__bf16)Wo[k * 128 + z];
}

// ---------------------------------------------------------------------------
// Kernel 0b: norm[i,j] = sum_s mask[s,i]*mask[s,j] + 1e-3
// ---------------------------------------------------------------------------
__global__ __launch_bounds__(256) void prep_norm_kernel(
    const float* __restrict__ mask, float* __restrict__ nrm)
{
    int idx = blockIdx.x * 256 + threadIdx.x;     // 65536 = 256*256
    int i = idx >> 8;
    int j = idx & 255;
    float acc = 0.f;
    for (int s = 0; s < 128; ++s)
        acc += mask[s * 256 + i] * mask[s * 256 + j];
    nrm[idx] = acc + 1e-3f;
}

// ---------------------------------------------------------------------------
// Kernel 1: LayerNorm + projections.
//   m [S=128, N=256, 256] f32 -> aT,bT [(n*32+c), s] bf16  (K-major for WMMA)
// One wave per (s,n) row; 8 rows per 256-thread block.
// ---------------------------------------------------------------------------
__global__ __launch_bounds__(256) void lnproj_kernel(
    const float* __restrict__ m, const float* __restrict__ mask,
    const float* __restrict__ g, const float* __restrict__ bvec,
    const float* __restrict__ Wa, const float* __restrict__ ba,
    const float* __restrict__ Wb, const float* __restrict__ bb,
    __bf16* __restrict__ aT, __bf16* __restrict__ bT)
{
    __shared__ float smem[8][256];
    const int lane = threadIdx.x & 31;
    const int warp = threadIdx.x >> 5;
    const int row  = blockIdx.x * 8 + warp;       // [0, 32768)
    const int s = row >> 8;
    const int n = row & 255;

    const float* mr = m + (size_t)row * 256;
    float x[8];
    float sum = 0.f, sq = 0.f;
#pragma unroll
    for (int t = 0; t < 8; ++t) {
        x[t] = mr[lane + 32 * t];
        sum += x[t];
        sq  += x[t] * x[t];
    }
#pragma unroll
    for (int off = 16; off > 0; off >>= 1) {
        sum += __shfl_xor(sum, off, 32);
        sq  += __shfl_xor(sq,  off, 32);
    }
    const float mu  = sum * (1.f / 256.f);
    const float var = sq * (1.f / 256.f) - mu * mu;
    const float rs  = rsqrtf(var + 1e-5f);

#pragma unroll
    for (int t = 0; t < 8; ++t) {
        int k = lane + 32 * t;
        smem[warp][k] = (x[t] - mu) * rs * g[k] + bvec[k];
    }
    __syncthreads();

    float aa = 0.f, bbv = 0.f;
#pragma unroll 8
    for (int k = 0; k < 256; ++k) {
        float v = smem[warp][k];
        aa  += v * Wa[k * 32 + lane];
        bbv += v * Wb[k * 32 + lane];
    }
    const float mk = mask[row];
    aa  = (aa  + ba[lane]) * mk;
    bbv = (bbv + bb[lane]) * mk;

    const size_t o = (size_t)(n * 32 + lane) * 128 + s;
    aT[o] = (__bf16)aa;
    bT[o] = (__bf16)bbv;
}

// ---------------------------------------------------------------------------
// Kernel 2 (fused): per block, 4x4 (i,j) pairs.
//   Stage A: outer tile 128x128 in (i,c)x(j,e) space, K=S=128, bf16 WMMA.
//   Stage B: project 16 pairs x K=1024 against WoT (bf16 WMMA), + bias, /norm.
// 256 threads = 8 waves; waves tiled 4x2 over the 128x128 outer tile.
// A/B tiles are pulled into LDS by the Tensor Data Mover when available.
// ---------------------------------------------------------------------------
__global__ __launch_bounds__(256) void opm_kernel(
    const __bf16* __restrict__ aT, const __bf16* __restrict__ bT,
    const __bf16* __restrict__ WoT, const float* __restrict__ normv,
    const float* __restrict__ bo, float* __restrict__ out)
{
    __shared__ union LdsU {
        struct { __bf16 A[128 * 136]; __bf16 B[128 * 136]; } ab; // stage A tiles
        __bf16 O[16 * 1032];                                     // stage B outer tile
    } lds;

    const int tid  = threadIdx.x;
    const int lane = tid & 31;
    const int warp = tid >> 5;
    const int r15  = lane & 15;
    const int hi   = lane >> 4;
    const int bi0  = blockIdx.x * 4;
    const int bj0  = blockIdx.y * 4;

    const __bf16* gA = aT + (size_t)blockIdx.x * (128 * 128);
    const __bf16* gB = bT + (size_t)blockIdx.y * (128 * 128);

    // warm L2 for this wave's WoT slab (stage B operand)
    const __bf16* wz = WoT + (size_t)(warp * 16 + r15) * 1024;
#pragma unroll
    for (int q = 0; q < 4; ++q)
        __builtin_prefetch(wz + q * 256 + hi * 128, 0, 1);

    // ---- load A/B tiles (each 128 rows x 128 bf16, contiguous 32KB) to LDS ----
#if defined(OPM_TDM)
    if (warp == 0) {
        unsigned ldsA = (unsigned)(size_t)(void*)&lds.ab.A[0]; // LDS byte offset
        unsigned ldsB = (unsigned)(size_t)(void*)&lds.ab.B[0];
        tdm_load_tile_2d(gA, ldsA);
        tdm_load_tile_2d(gB, ldsB);
    }
    __builtin_amdgcn_s_wait_tensorcnt(0);
    __syncthreads();
#elif defined(OPM_ASYNC)
#pragma unroll
    for (int t = 0; t < 8; ++t) {
        int linear = tid + t * 256;          // uint4 index, 2048 total
        int row  = linear >> 4;
        int col8 = (linear & 15) << 3;
        __builtin_amdgcn_global_load_async_to_lds_b128(
            (AS1 const void*)(gA + linear * 8),
            (AS3 void*)&lds.ab.A[row * 136 + col8], 0, 0);
        __builtin_amdgcn_global_load_async_to_lds_b128(
            (AS1 const void*)(gB + linear * 8),
            (AS3 void*)&lds.ab.B[row * 136 + col8], 0, 0);
    }
    __builtin_amdgcn_s_wait_asynccnt(0);
    __syncthreads();
#else
#pragma unroll
    for (int t = 0; t < 8; ++t) {
        int linear = tid + t * 256;          // uint4 index, 2048 total
        int row  = linear >> 4;              // 16 uint4 per 128-elem row
        int col8 = (linear & 15) << 3;       // bf16 column
        uint4 va = ((const uint4*)gA)[linear];
        uint4 vb = ((const uint4*)gB)[linear];
        *(uint4*)&lds.ab.A[row * 136 + col8] = va;
        *(uint4*)&lds.ab.B[row * 136 + col8] = vb;
    }
    __syncthreads();
#endif

    // ---- stage A: 8 accumulators of 16x16, K loop over S=128 in steps of 32 ----
    const int mbase = (warp >> 1) * 32;      // rows in (i,c) space
    const int nbase = (warp & 1) * 64;       // cols in (j,e) space
    v8f acc[2][4] = {};

#pragma unroll
    for (int ks = 0; ks < 4; ++ks) {
        const int k0 = ks * 32;
        Frag af[2];
#pragma unroll
        for (int mt = 0; mt < 2; ++mt) {
            int row = mbase + mt * 16 + r15;
            af[mt].h[0] = *(const v8bf*)&lds.ab.A[row * 136 + k0 + hi * 8];
            af[mt].h[1] = *(const v8bf*)&lds.ab.A[row * 136 + k0 + 16 + hi * 8];
        }
        Frag bf[4];
#pragma unroll
        for (int nt = 0; nt < 4; ++nt) {
            int col = nbase + nt * 16 + r15;
            bf[nt].h[0] = *(const v8bf*)&lds.ab.B[col * 136 + k0 + hi * 8];
            bf[nt].h[1] = *(const v8bf*)&lds.ab.B[col * 136 + k0 + 16 + hi * 8];
        }
#pragma unroll
        for (int mt = 0; mt < 2; ++mt)
#pragma unroll
            for (int nt = 0; nt < 4; ++nt)
                acc[mt][nt] = __builtin_amdgcn_wmma_f32_16x16x32_bf16(
                    false, af[mt].v, false, bf[nt].v,
                    (short)0, acc[mt][nt], false, false);
    }
    __syncthreads();   // stage-A reads done; O aliases A

    // ---- spill outer tile to LDS as [pair p(16) x ce(1024)] bf16 ----
#pragma unroll
    for (int mt = 0; mt < 2; ++mt)
#pragma unroll
        for (int nt = 0; nt < 4; ++nt)
#pragma unroll
            for (int r = 0; r < 8; ++r) {
                int R = mbase + mt * 16 + r + hi * 8;   // (i,c) row
                int C = nbase + nt * 16 + r15;          // (j,e) col
                int p = ((R >> 5) << 2) + (C >> 5);     // ii*4+jj
                lds.O[p * 1032 + (R & 31) * 32 + (C & 31)] = (__bf16)acc[mt][nt][r];
            }
    __syncthreads();

    // ---- stage B: [16 x 1024] @ WoT^T -> [16 x 128]; wave w owns z block w*16 ----
    v8f acc2 = {};
#pragma unroll 4
    for (int kk = 0; kk < 1024; kk += 32) {
        Frag af;
        af.h[0] = *(const v8bf*)&lds.O[r15 * 1032 + kk + hi * 16];
        af.h[1] = *(const v8bf*)&lds.O[r15 * 1032 + kk + hi * 16 + 8];
        Frag bf;
        bf.v = *(const v16bf*)&wz[kk + hi * 16];
        acc2 = __builtin_amdgcn_wmma_f32_16x16x32_bf16(
            false, af.v, false, bf.v, (short)0, acc2, false, false);
    }

    // ---- epilogue: + bias, / norm, scatter to out[i,j,z] ----
    const int z = warp * 16 + r15;
    const float boz = bo[z];
#pragma unroll
    for (int r = 0; r < 8; ++r) {
        int p = r + hi * 8;
        int i = bi0 + (p >> 2);
        int j = bj0 + (p & 3);
        float nv = normv[i * 256 + j];
        out[(size_t)(i * 256 + j) * 128 + z] = (acc2[r] + boz) / nv;
    }
}

// ---------------------------------------------------------------------------
extern "C" void kernel_launch(void* const* d_in, const int* in_sizes, int n_in,
                              void* d_out, int out_size, void* d_ws, size_t ws_size,
                              hipStream_t stream) {
    const float* m    = (const float*)d_in[0];
    const float* mask = (const float*)d_in[1];
    const float* ln_g = (const float*)d_in[2];
    const float* ln_b = (const float*)d_in[3];
    const float* Wa   = (const float*)d_in[4];
    const float* ba   = (const float*)d_in[5];
    const float* Wb   = (const float*)d_in[6];
    const float* bb   = (const float*)d_in[7];
    const float* Wo   = (const float*)d_in[8];
    const float* bo   = (const float*)d_in[9];
    float* out = (float*)d_out;

    char* ws = (char*)d_ws;
    __bf16* aT   = (__bf16*)(ws);                    // 8192*128*2 = 2 MB
    __bf16* bT   = (__bf16*)(ws + (2u << 20));       // 2 MB
    __bf16* WoT  = (__bf16*)(ws + (4u << 20));       // 256 KB
    float*  nrm  = (float*) (ws + (4u << 20) + (256u << 10)); // 256 KB

    prep_wot_kernel <<<512,  256, 0, stream>>>(Wo, WoT);
    prep_norm_kernel<<<256,  256, 0, stream>>>(mask, nrm);
    lnproj_kernel   <<<4096, 256, 0, stream>>>(m, mask, ln_g, ln_b,
                                               Wa, ba, Wb, bb, aT, bT);
    opm_kernel<<<dim3(64, 64), 256, 0, stream>>>(aT, bT, WoT, nrm, bo, out);
}
